// Encoder_30751965839570
// MI455X (gfx1250) — matrix operand order repaired and checked by reading, not compiled, not run on hardware
//
#include <hip/hip_runtime.h>
#include <hip/hip_bf16.h>

typedef __attribute__((ext_vector_type(16))) __bf16 v16bf;
typedef __attribute__((ext_vector_type(8)))  float  v8f;

#define N_NODES 50000
#define N_EDGES 800000
#define D_IN    512
#define D_H     512
#define D_C     512   // combined mu|logstd output width (256+256)

#ifndef __has_builtin
#define __has_builtin(x) 0
#endif

#if defined(__HIP_DEVICE_COMPILE__) && __has_builtin(__builtin_amdgcn_global_load_async_to_lds_b128)
#define HAVE_ASYNC_LDS 1
typedef __attribute__((__vector_size__(16))) int vi4_t;
typedef __attribute__((address_space(1))) vi4_t as1_vi4;
typedef __attribute__((address_space(3))) vi4_t as3_vi4;
#else
#define HAVE_ASYNC_LDS 0
#endif

__device__ __forceinline__ void wait_async_zero() {
#if HAVE_ASYNC_LDS
#if __has_builtin(__builtin_amdgcn_s_wait_asynccnt)
    __builtin_amdgcn_s_wait_asynccnt(0);
#else
    asm volatile("s_wait_asynccnt 0x0" ::: "memory");
#endif
#endif
}

// ------------------------------------------------------------------
// jax-compatible threefry2x32 (key = {0, 42} for jax.random.key(42))
// ------------------------------------------------------------------
__device__ __forceinline__ unsigned rotl32(unsigned x, unsigned r) {
    return (x << r) | (x >> (32u - r));
}

__device__ __forceinline__ void threefry2x32(unsigned k0, unsigned k1,
                                             unsigned x0, unsigned x1,
                                             unsigned* o0, unsigned* o1) {
    unsigned ks0 = k0, ks1 = k1, ks2 = k0 ^ k1 ^ 0x1BD11BDAu;
    x0 += ks0; x1 += ks1;
    const unsigned R[2][4] = {{13u,15u,26u,6u},{17u,29u,16u,24u}};
    unsigned ks[3] = {ks0, ks1, ks2};
    #pragma unroll
    for (int i = 0; i < 5; ++i) {
        #pragma unroll
        for (int j = 0; j < 4; ++j) {
            x0 += x1;
            x1 = rotl32(x1, R[i & 1][j]);
            x1 ^= x0;
        }
        x0 += ks[(i + 1) % 3];
        x1 += ks[(i + 2) % 3] + (unsigned)(i + 1);
    }
    *o0 = x0; *o1 = x1;
}

// ------------------------------------------------------------------
// small utility kernels
// ------------------------------------------------------------------
__global__ void fill_ones_kernel(float* p, int n) {
    int i = blockIdx.x * blockDim.x + threadIdx.x;
    if (i < n) p[i] = 1.0f;   // deg starts at 1.0 (self loop)
}

__global__ void zero_kernel(float* p, int n) {
    int i = blockIdx.x * blockDim.x + threadIdx.x;
    if (i < n) p[i] = 0.0f;
}

__global__ void count_deg_kernel(const long long* __restrict__ ei, float* deg, int E) {
    int e = blockIdx.x * blockDim.x + threadIdx.x;
    if (e < E) atomicAdd(&deg[ei[E + e]], 1.0f);   // dst = ei[E..2E)
}

__global__ void rsqrt_kernel(float* p, int n) {
    int i = blockIdx.x * blockDim.x + threadIdx.x;
    if (i < n) p[i] = rsqrtf(p[i]);   // in-place deg -> dinv
}

// dropout (train, p=0.5): keep iff threefry bits < 2^31; scale x by 2; emit bf16
__global__ void dropout_bf16_kernel(const float* __restrict__ x,
                                    __hip_bfloat16* __restrict__ h, int half_n) {
    int i = blockIdx.x * blockDim.x + threadIdx.x;
    if (i >= half_n) return;
    unsigned o0, o1;
    threefry2x32(0u, 42u, (unsigned)i, (unsigned)(i + half_n), &o0, &o1);
    float v0 = (o0 < 0x80000000u) ? x[i] * 2.0f : 0.0f;
    float v1 = (o1 < 0x80000000u) ? x[i + half_n] * 2.0f : 0.0f;
    h[i]          = __float2bfloat16(v0);
    h[i + half_n] = __float2bfloat16(v1);
}

// ------------------------------------------------------------------
// weight pre-pack into WMMA B-fragment layout (32x16 bf16 tiles)
// dword index: nt*4096 + kt*256 + lane*8 + v
//   lane: n = lane&15 (column within tile), half = lane>>4
//   dword v holds {B[kt*32 + half*16 + 2v][c], B[...+2v+1][c]}
// ------------------------------------------------------------------
__device__ __forceinline__ unsigned pack_bf16_pair(float lo, float hi) {
    __hip_bfloat16 a = __float2bfloat16(lo);
    __hip_bfloat16 b = __float2bfloat16(hi);
    unsigned short ua = *(unsigned short*)&a;
    unsigned short ub = *(unsigned short*)&b;
    return (unsigned)ua | ((unsigned)ub << 16);
}

__global__ void pack_w1_kernel(const float* __restrict__ W, unsigned* __restrict__ o) {
    int idx = blockIdx.x * blockDim.x + threadIdx.x;
    if (idx >= (D_IN * D_H) / 2) return;
    int v = idx & 7, lane = (idx >> 3) & 31, kt = (idx >> 8) & 15, nt = idx >> 12;
    int n = lane & 15, half = lane >> 4;
    int k = kt * 32 + half * 16 + v * 2;
    int c = nt * 16 + n;
    o[idx] = pack_bf16_pair(W[k * D_H + c], W[(k + 1) * D_H + c]);
}

__global__ void pack_wc_kernel(const float* __restrict__ Wmu, const float* __restrict__ Wls,
                               unsigned* __restrict__ o) {
    int idx = blockIdx.x * blockDim.x + threadIdx.x;
    if (idx >= (D_H * D_C) / 2) return;
    int v = idx & 7, lane = (idx >> 3) & 31, kt = (idx >> 8) & 15, nt = idx >> 12;
    int n = lane & 15, half = lane >> 4;
    int k = kt * 32 + half * 16 + v * 2;
    int c = nt * 16 + n;
    float lo, hi;
    if (c < 256) { lo = Wmu[k * 256 + c];         hi = Wmu[(k + 1) * 256 + c]; }
    else         { lo = Wls[k * 256 + (c - 256)]; hi = Wls[(k + 1) * 256 + (c - 256)]; }
    o[idx] = pack_bf16_pair(lo, hi);
}

// ------------------------------------------------------------------
// bf16 WMMA GEMM: C[M,NC] f32 = A[M,K] bf16 (row-major) @ Bpack (fragment layout)
// block tile 128x128, 8 wave32 waves, each wave 32x64 (2x4 16x16 tiles)
// A staged through LDS (async-to-LDS when available); B read directly as fragments
// ------------------------------------------------------------------
#define BM 128
#define BN 128
#define BK 32
#define A_STR 40   // padded LDS row stride (bf16), multiple of 8 -> 16B-aligned rows

__global__ __launch_bounds__(256)
void gemm_bf16_wmma(const __hip_bfloat16* __restrict__ A,
                    const unsigned* __restrict__ Bp,
                    float* __restrict__ C,
                    int M, int K, int NC) {
    __shared__ __hip_bfloat16 As[BM * A_STR];

    const int tid  = threadIdx.x;
    const int lane = tid & 31;
    const int wave = tid >> 5;
    const int wm   = wave & 3;   // 4 wave-rows * 32 rows
    const int wn   = wave >> 2;  // 2 wave-cols * 64 cols
    const int m_blk = blockIdx.x * BM;
    const int n_blk = blockIdx.y * BN;
    const bool full = (m_blk + BM) <= M;
    const int NKT = K >> 5;

    v8f acc[2][4] = {};

    const int mrow  = lane & 15;
    const int mhalf = lane >> 4;

    // A staging: 512 x 16B chunks, 2 per thread
    const int ar0 = tid >> 2,         aoff0 = (tid & 3) * 8;
    const int ar1 = (tid + 256) >> 2, aoff1 = ((tid + 256) & 3) * 8;

    // per-wave fragment bases
    const int afrag_row0 = wm * 32 + mrow;           // + mt*16
    int nt_g[4];
    #pragma unroll
    for (int nt = 0; nt < 4; ++nt) nt_g[nt] = (n_blk >> 4) + wn * 4 + nt;
    const uint4* __restrict__ bq = (const uint4*)Bp;

    for (int kt = 0; kt < NKT; ++kt) {
        const int k0 = kt * BK;
#if HAVE_ASYNC_LDS
        if (full) {
            __builtin_amdgcn_global_load_async_to_lds_b128(
                (as1_vi4*)(void*)(A + (size_t)(m_blk + ar0) * K + k0 + aoff0),
                (as3_vi4*)(void*)(As + ar0 * A_STR + aoff0), 0, 0);
            __builtin_amdgcn_global_load_async_to_lds_b128(
                (as1_vi4*)(void*)(A + (size_t)(m_blk + ar1) * K + k0 + aoff1),
                (as3_vi4*)(void*)(As + ar1 * A_STR + aoff1), 0, 0);
            wait_async_zero();
        } else
#endif
        {
            uint4 v0 = make_uint4(0u, 0u, 0u, 0u);
            uint4 v1 = make_uint4(0u, 0u, 0u, 0u);
            if (full || (m_blk + ar0) < M)
                v0 = *(const uint4*)(A + (size_t)(m_blk + ar0) * K + k0 + aoff0);
            if (full || (m_blk + ar1) < M)
                v1 = *(const uint4*)(A + (size_t)(m_blk + ar1) * K + k0 + aoff1);
            *(uint4*)(As + ar0 * A_STR + aoff0) = v0;
            *(uint4*)(As + ar1 * A_STR + aoff1) = v1;
        }
        __syncthreads();

        // A fragments (16x32 bf16, k = mhalf*16 + s, same order as Bpack): 2 x 16B contiguous per lane
        const unsigned short* as = (const unsigned short*)As;
        v16bf afrag[2];
        #pragma unroll
        for (int mt = 0; mt < 2; ++mt) {
            union { v16bf v; uint4 q[2]; } fa;
            const unsigned short* ap = as + (afrag_row0 + mt * 16) * A_STR + mhalf * 16;
            fa.q[0] = *(const uint4*)(ap);        // K = mhalf*16 + 0..7
            fa.q[1] = *(const uint4*)(ap + 8);    // K = mhalf*16 + 8..15
            afrag[mt] = fa.v;
        }
        // B fragments: pre-packed, 2 x b128 coalesced per lane
        v16bf bfrag[4];
        #pragma unroll
        for (int nt = 0; nt < 4; ++nt) {
            union { v16bf v; uint4 q[2]; } fb;
            const uint4* p = bq + (size_t)nt_g[nt] * 1024 + kt * 64 + lane * 2;
            fb.q[0] = p[0];
            fb.q[1] = p[1];
            bfrag[nt] = fb.v;
        }

        #pragma unroll
        for (int mt = 0; mt < 2; ++mt)
            #pragma unroll
            for (int nt = 0; nt < 4; ++nt)
                acc[mt][nt] = __builtin_amdgcn_wmma_f32_16x16x32_bf16(
                    false, afrag[mt], false, bfrag[nt],
                    (short)0, acc[mt][nt], false, false);

        __syncthreads();
    }

    // store: C/D layout: lane col = lane&15, vgpr r -> row r + 8*(lane>>4)
    #pragma unroll
    for (int mt = 0; mt < 2; ++mt) {
        const int gm0 = m_blk + wm * 32 + mt * 16 + mhalf * 8;
        #pragma unroll
        for (int nt = 0; nt < 4; ++nt) {
            const int gc = n_blk + wn * 64 + nt * 16 + mrow;
            float* cp = C + (size_t)gm0 * NC + gc;
            if (full) {
                #pragma unroll
                for (int r = 0; r < 8; ++r) { *cp = acc[mt][nt][r]; cp += NC; }
            } else {
                #pragma unroll
                for (int r = 0; r < 8; ++r) {
                    if ((gm0 + r) < M) *cp = acc[mt][nt][r];
                    cp += NC;
                }
            }
        }
    }
}

// ------------------------------------------------------------------
// edge aggregation: agg[dst] += xw[src] * (dinv[src]*dinv[dst])
// 128 threads per edge, float4 per thread, D fixed at 512
// ------------------------------------------------------------------
__global__ __launch_bounds__(256)
void scatter_edges_kernel(const long long* __restrict__ ei,
                          const float* __restrict__ dinv,
                          const float* __restrict__ xw,
                          float* __restrict__ agg, int E) {
    int gi = blockIdx.x * blockDim.x + threadIdx.x;
    if (gi >= E * 128) return;
    int e = gi >> 7;
    int q = gi & 127;
    long long s = ei[e];
    long long d = ei[E + e];
    if (q == 0) __builtin_prefetch(xw + (size_t)s * 512, 0, 0);
    float nrm = dinv[s] * dinv[d];
    const float4 v = *(const float4*)(xw + (size_t)s * 512 + q * 4);
    float* ap = agg + (size_t)d * 512 + q * 4;
    atomicAdd(ap + 0, v.x * nrm);
    atomicAdd(ap + 1, v.y * nrm);
    atomicAdd(ap + 2, v.z * nrm);
    atomicAdd(ap + 3, v.w * nrm);
}

// layer-1 epilogue: relu(agg + xw*self_norm + b1) -> bf16
__global__ void finalize1_kernel(const float* __restrict__ agg,
                                 const float* __restrict__ xw,
                                 const float* __restrict__ dinv,
                                 const float* __restrict__ b1,
                                 __hip_bfloat16* __restrict__ h1, int total) {
    int i = blockIdx.x * blockDim.x + threadIdx.x;
    if (i >= total) return;
    int n = i >> 9, c = i & 511;
    float sn = dinv[n]; sn *= sn;
    float v = agg[i] + xw[i] * sn + b1[c];
    h1[i] = __float2bfloat16(v > 0.0f ? v : 0.0f);
}

// layer-2 epilogue: split combined 512-wide result into mu / logstd regions of d_out
__global__ void finalize2_kernel(const float* __restrict__ agg,
                                 const float* __restrict__ xw,
                                 const float* __restrict__ dinv,
                                 const float* __restrict__ bmu,
                                 const float* __restrict__ bls,
                                 float* __restrict__ out, int total) {
    int i = blockIdx.x * blockDim.x + threadIdx.x;
    if (i >= total) return;
    int n = i >> 9, c = i & 511;
    float sn = dinv[n]; sn *= sn;
    float v = agg[i] + xw[i] * sn + ((c < 256) ? bmu[c] : bls[c - 256]);
    if (c < 256) out[(size_t)n * 256 + c] = v;
    else         out[(size_t)N_NODES * 256 + (size_t)n * 256 + (c - 256)] = v;
}

// ------------------------------------------------------------------
// launch
// ------------------------------------------------------------------
extern "C" void kernel_launch(void* const* d_in, const int* in_sizes, int n_in,
                              void* d_out, int out_size, void* d_ws, size_t ws_size,
                              hipStream_t stream) {
    const float*     x   = (const float*)d_in[0];
    const long long* ei  = (const long long*)d_in[1];   // [2,E] int64
    const float*     W1  = (const float*)d_in[2];
    const float*     b1  = (const float*)d_in[3];
    const float*     Wmu = (const float*)d_in[4];
    const float*     bmu = (const float*)d_in[5];
    const float*     Wls = (const float*)d_in[6];
    const float*     bls = (const float*)d_in[7];
    float* out = (float*)d_out;

    // workspace carve (~308 MB)
    char* p = (char*)d_ws;
    auto take = [&](size_t bytes) -> char* {
        char* r = p;
        p += (bytes + 255) & ~(size_t)255;
        return r;
    };
    float*          dinv  = (float*)take((size_t)N_NODES * 4);
    __hip_bfloat16* hdrop = (__hip_bfloat16*)take((size_t)N_NODES * D_IN * 2);
    unsigned*       w1p   = (unsigned*)take((size_t)(D_IN * D_H) / 2 * 4);
    unsigned*       wcp   = (unsigned*)take((size_t)(D_H * D_C) / 2 * 4);
    float*          xw    = (float*)take((size_t)N_NODES * D_H * 4);   // reused layer2
    float*          agg   = (float*)take((size_t)N_NODES * D_H * 4);   // reused layer2
    __hip_bfloat16* h1bf  = (__hip_bfloat16*)take((size_t)N_NODES * D_H * 2);

    const int TPB = 256;
    // degrees -> dinv
    fill_ones_kernel<<<(N_NODES + TPB - 1) / TPB, TPB, 0, stream>>>(dinv, N_NODES);
    count_deg_kernel<<<(N_EDGES + TPB - 1) / TPB, TPB, 0, stream>>>(ei, dinv, N_EDGES);
    rsqrt_kernel<<<(N_NODES + TPB - 1) / TPB, TPB, 0, stream>>>(dinv, N_NODES);

    // dropout -> bf16 activations
    int half_n = N_NODES * D_IN / 2;
    dropout_bf16_kernel<<<(half_n + TPB - 1) / TPB, TPB, 0, stream>>>(x, hdrop, half_n);

    // weights -> WMMA B-fragment layout (Wmu|Wls fused into one 512-wide matrix)
    pack_w1_kernel<<<((D_IN * D_H / 2) + TPB - 1) / TPB, TPB, 0, stream>>>(W1, w1p);
    pack_wc_kernel<<<((D_H * D_C / 2) + TPB - 1) / TPB, TPB, 0, stream>>>(Wmu, Wls, wcp);

    dim3 ggrid((N_NODES + BM - 1) / BM, D_H / BN);

    // layer 1: GEMM -> aggregate -> relu/bias -> bf16
    gemm_bf16_wmma<<<ggrid, 256, 0, stream>>>(hdrop, w1p, xw, N_NODES, D_IN, D_H);
    int tot = N_NODES * D_H;
    zero_kernel<<<(tot + TPB - 1) / TPB, TPB, 0, stream>>>(agg, tot);
    scatter_edges_kernel<<<(N_EDGES * 128 + TPB - 1) / TPB, TPB, 0, stream>>>(ei, dinv, xw, agg, N_EDGES);
    finalize1_kernel<<<(tot + TPB - 1) / TPB, TPB, 0, stream>>>(agg, xw, dinv, b1, h1bf, tot);

    // layer 2 (mu|logstd fused): GEMM -> aggregate -> bias/split
    gemm_bf16_wmma<<<ggrid, 256, 0, stream>>>(h1bf, wcp, xw, N_NODES, D_H, D_C);
    zero_kernel<<<(tot + TPB - 1) / TPB, TPB, 0, stream>>>(agg, tot);
    scatter_edges_kernel<<<(N_EDGES * 128 + TPB - 1) / TPB, TPB, 0, stream>>>(ei, dinv, xw, agg, N_EDGES);
    finalize2_kernel<<<(tot + TPB - 1) / TPB, TPB, 0, stream>>>(agg, xw, dinv, bmu, bls, out, tot);
}
